// CausalSelfAttention_49520972923304
// MI455X (gfx1250) — matrix operand order repaired and checked by reading, not compiled, not run on hardware
//
#include <hip/hip_runtime.h>

// ---------------------------------------------------------------------------
// Problem constants (match reference: DIM=1024, H=16, d=64, B=2, T=2048)
// ---------------------------------------------------------------------------
static constexpr int kB    = 2;
static constexpr int kT    = 2048;
static constexpr int kC    = 1024;
static constexpr int kH    = 16;
static constexpr int kHd   = 64;          // head dim
static constexpr int kBT   = kB * kT;     // 4096 rows
static constexpr int k3C   = 3 * kC;      // 3072

// ---------------------------------------------------------------------------
// CDNA5 WMMA types
// ---------------------------------------------------------------------------
typedef __attribute__((ext_vector_type(16))) __bf16 v16bf;
typedef __attribute__((ext_vector_type(8)))  float  v8f;

typedef unsigned int u32x4 __attribute__((ext_vector_type(4)));
typedef int          i32x8 __attribute__((ext_vector_type(8)));
typedef int          i32x4 __attribute__((ext_vector_type(4)));

union AFrag {            // 16 bf16 = 8 VGPRs, matches v_wmma A/B operand
    v16bf          v;
    unsigned short u[16];
};

__device__ __forceinline__ unsigned short f32_to_bf16(float f) {
    unsigned int x = __float_as_uint(f);
    unsigned int rnd = 0x7FFFu + ((x >> 16) & 1u);   // round-to-nearest-even
    return (unsigned short)((x + rnd) >> 16);
}

// ---------------------------------------------------------------------------
// Tensor Data Mover: 2-D tile load (2-byte elements) global -> LDS.
// Descriptor per CDNA5 ISA.
// pad_interval_code: pad every 2<<code DWORDs; pad_amount_code: code+1 DWORDs.
// Builtin arity: 6 args on this toolchain (clang-23 / therock headers).
// ---------------------------------------------------------------------------
__device__ __forceinline__ void tdm_load_tile_2d(
    unsigned lds_addr, const void* gptr,
    unsigned tile_w_elems, unsigned tile_h,
    unsigned long long row_stride_elems,
    unsigned pad_interval_code, unsigned pad_amount_code) {
    unsigned long long ga = (unsigned long long)(size_t)gptr;
    u32x4 g0;
    g0[0] = 1u;                                   // count=1 valid, load, no gather
    g0[1] = lds_addr;                             // LDS byte address
    g0[2] = (unsigned)(ga & 0xFFFFFFFFu);         // global_addr[31:0]
    g0[3] = (unsigned)((ga >> 32) & 0x01FFFFFFu)  // global_addr[56:32]
          | 0x80000000u;                          // type=2 ("image")
    i32x8 g1;
    g1[0] = (int)((1u << 16)                      // data_size: 2 bytes
          | (1u << 20)                            // pad_enable
          | (pad_interval_code << 22)
          | (pad_amount_code << 25));
    g1[1] = (int)((tile_w_elems & 0xFFFFu) << 16);            // tensor_dim0[15:0]
    g1[2] = (int)(((tile_w_elems >> 16) & 0xFFFFu)            // tensor_dim0[31:16]
          | ((tile_h & 0xFFFFu) << 16));                      // tensor_dim1[15:0]
    g1[3] = (int)(((tile_h >> 16) & 0xFFFFu)                  // tensor_dim1[31:16]
          | ((tile_w_elems & 0xFFFFu) << 16));                // tile_dim0
    g1[4] = (int)(tile_h & 0xFFFFu);                          // tile_dim1 (dim2=0)
    g1[5] = (int)(unsigned)(row_stride_elems & 0xFFFFFFFFull);// dim0_stride lo32
    g1[6] = (int)(unsigned)((row_stride_elems >> 32) & 0xFFFFull);
    g1[7] = 0;
    i32x4 z4 = {0, 0, 0, 0};
    i32x8 z8 = {0, 0, 0, 0, 0, 0, 0, 0};
    __builtin_amdgcn_tensor_load_to_lds(g0, g1, z4, z4, z8, 0);
}

__device__ __forceinline__ unsigned lds_offset_of(const void* p) {
    return (unsigned)(size_t)p;                   // low 32 bits = LDS byte offset
}

// ---------------------------------------------------------------------------
// Kernel 0: fp32 -> bf16 conversion
// ---------------------------------------------------------------------------
__global__ void cvt_f32_bf16(const float* __restrict__ in,
                             unsigned short* __restrict__ out, int n) {
    int i = blockIdx.x * blockDim.x + threadIdx.x;
    if (i < n) out[i] = f32_to_bf16(in[i]);
}

// ---------------------------------------------------------------------------
// Kernel 1/3: C[M,N] = A[M,K] * W[N,K]^T   (bf16 in, fp32 accumulate)
// Block tile 64x64, K-step 32; 128 threads = 4 waves, each wave 32x32
// (2x2 of 16x16x32 WMMA tiles). Tiles staged by TDM, double-buffered on
// TENSORcnt so the next DMA overlaps compute.
// ---------------------------------------------------------------------------
#define GBM 64
#define GBN 64
#define GBK 32
#define GPAD 16   // 16 bf16 = 32 B = 8 DWORDs of per-row pad

template <int STORE_BF16>
__global__ __launch_bounds__(128)
void gemm_xWT_wmma(const unsigned short* __restrict__ A,
                   const unsigned short* __restrict__ W,
                   void* __restrict__ Cout,
                   int M, int N, int K) {
    __shared__ unsigned short lA[2][GBM][GBK + GPAD];
    __shared__ unsigned short lB[2][GBN][GBK + GPAD];

    const int tid  = threadIdx.x;
    const int lane = tid & 31;
    const int wave = tid >> 5;
    const int hf   = lane >> 4;      // lane half (wave32)
    const int l16  = lane & 15;
    const int m0   = blockIdx.y * GBM;
    const int n0   = blockIdx.x * GBN;
    const int wm   = (wave >> 1) * 32;
    const int wn   = (wave & 1) * 32;

    v8f acc[2][2];
#pragma unroll
    for (int i = 0; i < 2; ++i)
#pragma unroll
        for (int j = 0; j < 2; ++j)
#pragma unroll
            for (int r = 0; r < 8; ++r) acc[i][j][r] = 0.0f;

    const int niter = K / GBK;
    // preamble: DMA first tiles into buffer 0
    if (wave == 0) {
        // 32-elem rows = 16 DWORDs -> interval code 3; pad 8 DWORDs -> code 7
        tdm_load_tile_2d(lds_offset_of(&lA[0][0][0]),
                         &A[(size_t)m0 * K], GBK, GBM, (unsigned long long)K, 3, 7);
        tdm_load_tile_2d(lds_offset_of(&lB[0][0][0]),
                         &W[(size_t)n0 * K], GBK, GBN, (unsigned long long)K, 3, 7);
    }

    for (int it = 0; it < niter; ++it) {
        const int c = it & 1;
        if (wave == 0) {
            if (it + 1 < niter) {
                const int kn = (it + 1) * GBK;
                tdm_load_tile_2d(lds_offset_of(&lA[1 - c][0][0]),
                                 &A[(size_t)m0 * K + kn], GBK, GBM,
                                 (unsigned long long)K, 3, 7);
                tdm_load_tile_2d(lds_offset_of(&lB[1 - c][0][0]),
                                 &W[(size_t)n0 * K + kn], GBK, GBN,
                                 (unsigned long long)K, 3, 7);
                // 2 newly issued outstanding; previous pair (buffer c) done
                __builtin_amdgcn_s_wait_tensorcnt(2);
            } else {
                __builtin_amdgcn_s_wait_tensorcnt(0);
            }
        }
        __syncthreads();               // buffer c ready for all waves

        // build fragments: contiguous k = 16*half + i, same order for A and B
        AFrag af[2], bfr[2];
#pragma unroll
        for (int t = 0; t < 2; ++t) {
            const int arow = wm + t * 16 + l16;   // A: M = lane&15
            const int brow = wn + t * 16 + l16;   // B: N = lane&15
#pragma unroll
            for (int i = 0; i < 16; ++i) {
                int ka = (hf ? 16 : 0) + i;                    // A K-halves
                int kb = (hf ? 16 : 0) + i;                    // B K-halves
                af[t].u[i]  = lA[c][arow][ka];
                bfr[t].u[i] = lB[c][brow][kb];
            }
        }
#pragma unroll
        for (int ti = 0; ti < 2; ++ti)
#pragma unroll
            for (int tj = 0; tj < 2; ++tj)
                acc[ti][tj] = __builtin_amdgcn_wmma_f32_16x16x32_bf16(
                    false, af[ti].v, false, bfr[tj].v,
                    (short)0, acc[ti][tj], false, false);

        __syncthreads();               // reads of buffer c done before refill
    }

    // epilogue: C/D layout row = r + 8*half, col = lane&15
#pragma unroll
    for (int ti = 0; ti < 2; ++ti)
#pragma unroll
        for (int tj = 0; tj < 2; ++tj)
#pragma unroll
            for (int r = 0; r < 8; ++r) {
                int row = m0 + wm + ti * 16 + r + 8 * hf;
                int col = n0 + wn + tj * 16 + l16;
                float val = acc[ti][tj][r];
                if (STORE_BF16)
                    ((unsigned short*)Cout)[(size_t)row * N + col] = f32_to_bf16(val);
                else
                    ((float*)Cout)[(size_t)row * N + col] = val;
            }
}

// ---------------------------------------------------------------------------
// Kernel 2: causal flash attention (bf16 WMMA, fp32 online softmax)
// grid = (T/64, B*H); 128 threads = 4 waves; wave owns 16 query rows.
// K/V tiles staged by TDM. qkv: [B*T, 3C] bf16. outb: [B*T, C] bf16.
// ---------------------------------------------------------------------------
#define AT_BQ 64
#define AT_BK 32

__global__ __launch_bounds__(128)
void flash_attn_wmma(const unsigned short* __restrict__ qkv,
                     unsigned short* __restrict__ outb) {
    __shared__ unsigned short lK[AT_BK][kHd + 8];
    __shared__ unsigned short lV[AT_BK][kHd + 8];
    __shared__ unsigned short lP[4][16][AT_BK + 8];

    const int tid  = threadIdx.x;
    const int lane = tid & 31;
    const int wave = tid >> 5;
    const int hf   = lane >> 4;
    const int l16  = lane & 15;

    const int q0 = blockIdx.x * AT_BQ;
    const int bh = blockIdx.y;
    const int b  = bh / kH;
    const int h  = bh % kH;

    const size_t rs   = (size_t)k3C;               // qkv row stride (elems)
    const size_t base = (size_t)b * kT * rs;
    const int qcol = h * kHd;
    const int kcol = kC + h * kHd;
    const int vcol = 2 * kC + h * kHd;

    // Q A-fragments for the two 32-wide d-chunks (loaded once)
    AFrag qf[2];
    {
        const int qrow = q0 + wave * 16 + l16;     // A: M = lane&15
        const unsigned short* qp = &qkv[base + (size_t)qrow * rs + qcol];
#pragma unroll
        for (int c = 0; c < 2; ++c)
#pragma unroll
            for (int i = 0; i < 16; ++i) {
                int kk = (hf ? 16 : 0) + i + c * 32;
                qf[c].u[i] = qp[kk];
            }
    }

    v8f oacc[4];
#pragma unroll
    for (int j = 0; j < 4; ++j)
#pragma unroll
        for (int r = 0; r < 8; ++r) oacc[j][r] = 0.0f;

    float mstate[8], lstate[8];
#pragma unroll
    for (int r = 0; r < 8; ++r) { mstate[r] = -3.0e38f; lstate[r] = 0.0f; }

    const int nkt = (q0 + AT_BQ) / AT_BK;          // causal: only tiles <= q range
    for (int kt = 0; kt < nkt; ++kt) {
        const int k0 = kt * AT_BK;
        __syncthreads();                           // prior reads of lK/lV done
        if (wave == 0) {
            // 64-elem rows = 32 DWORDs -> interval code 4; pad 4 DWORDs -> code 3
            tdm_load_tile_2d(lds_offset_of(&lK[0][0]),
                             &qkv[base + (size_t)k0 * rs + kcol],
                             kHd, AT_BK, (unsigned long long)rs, 4, 3);
            tdm_load_tile_2d(lds_offset_of(&lV[0][0]),
                             &qkv[base + (size_t)k0 * rs + vcol],
                             kHd, AT_BK, (unsigned long long)rs, 4, 3);
            __builtin_amdgcn_s_wait_tensorcnt(0);
        }
        __syncthreads();                           // tiles visible to all waves

        // S = Q * K^T (two 16-key tiles, accumulate over two d-chunks)
        v8f sacc[2];
#pragma unroll
        for (int t = 0; t < 2; ++t) {
#pragma unroll
            for (int r = 0; r < 8; ++r) sacc[t][r] = 0.0f;
#pragma unroll
            for (int c = 0; c < 2; ++c) {
                AFrag bk;                          // B: column n = key (lane&15)
#pragma unroll
                for (int i = 0; i < 16; ++i)
                    bk.u[i] = lK[t * 16 + l16][c * 32 + (hf ? 16 : 0) + i];
                sacc[t] = __builtin_amdgcn_wmma_f32_16x16x32_bf16(
                    false, qf[c].v, false, bk.v, (short)0, sacc[t], false, false);
            }
        }

        // scale + causal mask
        float s[2][8];
#pragma unroll
        for (int t = 0; t < 2; ++t)
#pragma unroll
            for (int r = 0; r < 8; ++r) {
                int qrow = q0 + wave * 16 + r + 8 * hf;
                int kc   = k0 + t * 16 + l16;
                float x  = sacc[t][r] * 0.125f;    // 1/sqrt(64)
                s[t][r]  = (kc <= qrow) ? x : -3.0e38f;
            }

        // fp32 online softmax (row reductions across the 16-lane half)
        float pr[2][8];
#pragma unroll
        for (int r = 0; r < 8; ++r) {
            float rmax = fmaxf(s[0][r], s[1][r]);
#pragma unroll
            for (int m = 1; m <= 8; m <<= 1)
                rmax = fmaxf(rmax, __shfl_xor(rmax, m, 32));
            float mnew  = fmaxf(mstate[r], rmax);
            float alpha = __expf(mstate[r] - mnew);
            float p0 = __expf(s[0][r] - mnew);
            float p1 = __expf(s[1][r] - mnew);
            float ps = p0 + p1;
#pragma unroll
            for (int m = 1; m <= 8; m <<= 1)
                ps += __shfl_xor(ps, m, 32);
            lstate[r] = lstate[r] * alpha + ps;
            mstate[r] = mnew;
#pragma unroll
            for (int j = 0; j < 4; ++j) oacc[j][r] *= alpha;
            pr[0][r] = p0; pr[1][r] = p1;
        }

        // P: C-layout -> LDS -> A-layout fragment
#pragma unroll
        for (int t = 0; t < 2; ++t)
#pragma unroll
            for (int r = 0; r < 8; ++r)
                lP[wave][r + 8 * hf][t * 16 + l16] = f32_to_bf16(pr[t][r]);
        __syncthreads();

        AFrag pf;
#pragma unroll
        for (int i = 0; i < 16; ++i) {
            int kk = (hf ? 16 : 0) + i;
            pf.u[i] = lP[wave][l16][kk];
        }

        // O += P * V  (4 output chunks of 16 cols)
#pragma unroll
        for (int j = 0; j < 4; ++j) {
            AFrag vf;                              // B: column n = d (lane&15)
#pragma unroll
            for (int i = 0; i < 16; ++i)
                vf.u[i] = lV[(hf ? 16 : 0) + i][j * 16 + l16];
            oacc[j] = __builtin_amdgcn_wmma_f32_16x16x32_bf16(
                false, pf.v, false, vf.v, (short)0, oacc[j], false, false);
        }
    }

    // normalize + store [b*T + q, h*64 + d] bf16
#pragma unroll
    for (int j = 0; j < 4; ++j)
#pragma unroll
        for (int r = 0; r < 8; ++r) {
            int row = q0 + wave * 16 + r + 8 * hf;
            int col = h * kHd + j * 16 + l16;
            float v = oacc[j][r] / lstate[r];
            outb[(size_t)(b * kT + row) * kC + col] = f32_to_bf16(v);
        }
}

// ---------------------------------------------------------------------------
// Host: x -> bf16 -> QKV gemm -> flash attn -> proj gemm -> d_out (fp32)
// ---------------------------------------------------------------------------
extern "C" void kernel_launch(void* const* d_in, const int* in_sizes, int n_in,
                              void* d_out, int out_size, void* d_ws, size_t ws_size,
                              hipStream_t stream) {
    const float* x     = (const float*)d_in[0];   // [B,T,C]
    const float* Wqkv  = (const float*)d_in[1];   // [3C,C]
    const float* Wproj = (const float*)d_in[2];   // [C,C]
    float* out = (float*)d_out;                   // [B,T,C]

    char* ws = (char*)d_ws;
    size_t off = 0;
    auto carve = [&](size_t bytes) -> void* {
        void* p = ws + off;
        off = (off + bytes + 255) & ~(size_t)255;
        return p;
    };
    unsigned short* xb     = (unsigned short*)carve((size_t)kBT * kC * 2);
    unsigned short* wqkvb  = (unsigned short*)carve((size_t)k3C * kC * 2);
    unsigned short* wprojb = (unsigned short*)carve((size_t)kC * kC * 2);
    unsigned short* qkvb   = (unsigned short*)carve((size_t)kBT * k3C * 2);
    unsigned short* attnb  = (unsigned short*)carve((size_t)kBT * kC * 2);
    (void)ws_size; (void)in_sizes; (void)n_in; (void)out_size;

    // 0) fp32 -> bf16
    {
        int n0 = kBT * kC, n1 = k3C * kC, n2 = kC * kC;
        cvt_f32_bf16<<<(n0 + 255) / 256, 256, 0, stream>>>(x, xb, n0);
        cvt_f32_bf16<<<(n1 + 255) / 256, 256, 0, stream>>>(Wqkv, wqkvb, n1);
        cvt_f32_bf16<<<(n2 + 255) / 256, 256, 0, stream>>>(Wproj, wprojb, n2);
    }
    // 1) qkv = x @ Wqkv^T   [4096 x 3072]
    {
        dim3 grid(k3C / GBN, kBT / GBM);
        gemm_xWT_wmma<1><<<grid, 128, 0, stream>>>(xb, wqkvb, qkvb, kBT, k3C, kC);
    }
    // 2) causal flash attention -> attnb [4096 x 1024]
    {
        dim3 grid(kT / AT_BQ, kB * kH);
        flash_attn_wmma<<<grid, 128, 0, stream>>>(qkvb, attnb);
    }
    // 3) out = attn @ Wproj^T  (fp32 epilogue straight to d_out)
    {
        dim3 grid(kC / GBN, kBT / GBM);
        gemm_xWT_wmma<0><<<grid, 128, 0, stream>>>(attnb, wprojb, out, kBT, kC, kC);
    }
}
